// FedFIMModel_84026740179407
// MI455X (gfx1250) — compile-verified, matrix operand order checked by reading, not run on hardware
//
#include <hip/hip_runtime.h>

// ---------------- dims ----------------
#define CB 32768   // batch
#define CC 20      // clients
#define SE 64      // price seq len
#define FP 8       // price feats
#define DSNT 64    // sentiment dim
#define DBH 32     // behavior dim
#define EE 256     // embed dim
#define DF 512     // fusion dim
#define HB 256     // adapter bottleneck
#define HHD 256    // head hidden
#define NOUT 8     // NC + 1 + AD

typedef __bf16 bf16_t;
typedef __attribute__((ext_vector_type(16))) __bf16 v16bf;
typedef __attribute__((ext_vector_type(8)))  float  v8f;

#define DEV __device__ __forceinline__

// A/B fragment for v_wmma_f32_16x16x32_bf16 (wave32):
// lane L holds row (L&15); K elements: elems 0..7 -> K = ks + (L>>4)*8 + j,
// elems 8..15 -> K = ks + 16 + (L>>4)*8 + j.  (B identical with column = L&15,
// weights stored transposed [N][Kpad] so rows are contiguous in K.)
DEV v16bf ldA_bf(const bf16_t* rp, int ks, int hi8) {
  const bf16_t* p = rp + ks + hi8;
  v16bf a;
#pragma unroll
  for (int j = 0; j < 8; ++j) { a[j] = p[j]; a[j + 8] = p[j + 16]; }
  return a;
}
DEV v8f wmma_bf(v16bf a, v16bf b, v8f c) {
  return __builtin_amdgcn_wmma_f32_16x16x32_bf16(false, a, false, b, (short)0, c,
                                                 false, false);
}

// ------------- prep: transpose+convert weights to bf16 [batch][N][Kpad] -----
__global__ void k_convT(bf16_t* dst, const float* src, int batch, int K, int N,
                        int Kpad) {
  long total = (long)batch * N * Kpad;
  for (long i = blockIdx.x * (long)blockDim.x + threadIdx.x; i < total;
       i += (long)gridDim.x * blockDim.x) {
    int k = (int)(i % Kpad);
    long r = i / Kpad;
    int n = (int)(r % N);
    int c = (int)(r / N);
    dst[i] = (k < K) ? (bf16_t)src[((long)c * K + k) * N + n] : (bf16_t)0.f;
  }
}

// pack Wd[C,256,3] | Wr[C,256] | Wa[C,256,4] into headT[C][16][256] (bf16)
__global__ void k_headpack(bf16_t* headT, float* headB, const float* Wd,
                           const float* bd, const float* Wr, const float* br,
                           const float* Wa, const float* ba) {
  int i0 = blockIdx.x * blockDim.x + threadIdx.x;
  int stride = gridDim.x * blockDim.x;
  for (int idx = i0; idx < CC * 16 * 256; idx += stride) {
    int k = idx & 255, n = (idx >> 8) & 15, c = idx >> 12;
    float v = 0.f;
    if (n < 3)       v = Wd[((long)c * 256 + k) * 3 + n];
    else if (n == 3) v = Wr[(long)c * 256 + k];
    else if (n < 8)  v = Wa[((long)c * 256 + k) * 4 + (n - 4)];
    headT[idx] = (bf16_t)v;
  }
  for (int idx = i0; idx < CC * 16; idx += stride) {
    int n = idx & 15, c = idx >> 4;
    float v = 0.f;
    if (n < 3)       v = bd[c * 3 + n];
    else if (n == 3) v = br[c];
    else if (n < 8)  v = ba[c * 4 + (n - 4)];
    headB[idx] = v;
  }
}

// ------------- counting sort by client id -------------
__global__ void k_zero(int* counts, int* cursor) {
  int i = threadIdx.x;
  if (i < CC) { counts[i] = 0; cursor[i] = 0; }
}
__global__ void k_count(const int* cid, int* counts) {
  int i = blockIdx.x * blockDim.x + threadIdx.x;
  if (i < CB) atomicAdd(&counts[cid[i]], 1);
}
__global__ void k_scan(const int* counts, int* offs, int* tstart) {
  if (threadIdx.x == 0 && blockIdx.x == 0) {
    int run = 0, tr = 0;
    for (int c = 0; c < CC; ++c) {
      offs[c] = run;
      tstart[c] = tr;
      run += counts[c];
      tr += (counts[c] + 15) >> 4;
    }
    offs[CC] = run;
    tstart[CC] = tr;
  }
}
__global__ void k_place(const int* cid, const int* offs, int* cursor, int* perm) {
  int i = blockIdx.x * blockDim.x + threadIdx.x;
  if (i < CB) {
    int c = cid[i];
    int p = atomicAdd(&cursor[c], 1);
    perm[offs[c] + p] = i;
  }
}

// ------------- trunk: encoders + attention fusion -> fused[B,512] -------------
__global__ __launch_bounds__(128) void k_encode_fuse(
    const float* __restrict__ price, const float* __restrict__ sent,
    const float* __restrict__ behav, const bf16_t* __restrict__ encPT,
    const float* __restrict__ encPb, const bf16_t* __restrict__ encST,
    const float* __restrict__ encSb, const bf16_t* __restrict__ encBT,
    const float* __restrict__ encBb, const bf16_t* __restrict__ attnWT,
    const float* __restrict__ attnV, const bf16_t* __restrict__ fuseWT,
    const float* __restrict__ fuseB, float* __restrict__ fused) {
  __shared__ bf16_t s_mp[16][32];        // price mean, K-padded to 32
  __shared__ bf16_t s_sent[16][DSNT];    // staged sentiment (bf16)
  __shared__ bf16_t s_beh[16][DBH];      // staged behavior (bf16)
  __shared__ bf16_t s_emb[3][16][EE];    // 3 modality embeddings (relu'd)
  __shared__ float s_sc[16][3];
  __shared__ float s_w[16][3];
  __shared__ bf16_t s_wemb[16][EE];      // attention-weighted embedding

  const int tid = threadIdx.x;
  const int lane = tid & 31;
  const int wv = tid >> 5;               // 4 waves
  const int col = lane & 15;
  const int hi8 = (lane >> 4) * 8;
  const int rsel = (lane >> 4) << 3;     // C/D row offset for this lane group
  const int rowg0 = blockIdx.x * 16;

  for (int i = tid; i < 16 * 32; i += 128) ((bf16_t*)s_mp)[i] = (bf16_t)0.f;
  if (tid < 48) ((float*)s_sc)[tid] = 0.f;
  __syncthreads();

  // temporal mean over price sequence + stage sentiment/behavior as bf16
  {
    int r = tid >> 3, f = tid & 7;       // 128 threads = 16 rows x 8 feats
    float s = 0.f;
    const float* p = price + (size_t)(rowg0 + r) * (SE * FP) + f;
#pragma unroll 8
    for (int t = 0; t < SE; ++t) s += p[t * FP];
    s_mp[r][f] = (bf16_t)(s * (1.f / SE));
  }
  for (int i = tid; i < 16 * DSNT; i += 128) {
    int r = i >> 6, n = i & 63;
    s_sent[r][n] = (bf16_t)sent[(size_t)(rowg0 + r) * DSNT + n];
  }
  for (int i = tid; i < 16 * DBH; i += 128) {
    int r = i >> 5, n = i & 31;
    s_beh[r][n] = (bf16_t)behav[(size_t)(rowg0 + r) * DBH + n];
  }
  __syncthreads();

  // encoders: e_m = relu(x_m @ W_m + b_m), 16 N-tiles split over 4 waves
  for (int nt = wv * 4; nt < wv * 4 + 4; ++nt) {
    int nb = nt * 16;
    // price (K=8 padded to 32)
    {
      v8f acc; float bb = encPb[nb + col];
#pragma unroll
      for (int r = 0; r < 8; ++r) acc[r] = bb;
      acc = wmma_bf(ldA_bf(&s_mp[col][0], 0, hi8),
                    ldA_bf(encPT + (size_t)(nb + col) * 32, 0, hi8), acc);
#pragma unroll
      for (int r = 0; r < 8; ++r) {
        float v = acc[r];
        s_emb[0][r + rsel][nb + col] = (bf16_t)(v > 0.f ? v : 0.f);
      }
    }
    // sentiment (K=64)
    {
      v8f acc; float bb = encSb[nb + col];
#pragma unroll
      for (int r = 0; r < 8; ++r) acc[r] = bb;
      const bf16_t* bp = encST + (size_t)(nb + col) * 64;
#pragma unroll
      for (int ks = 0; ks < 64; ks += 32)
        acc = wmma_bf(ldA_bf(&s_sent[col][0], ks, hi8), ldA_bf(bp, ks, hi8), acc);
#pragma unroll
      for (int r = 0; r < 8; ++r) {
        float v = acc[r];
        s_emb[1][r + rsel][nb + col] = (bf16_t)(v > 0.f ? v : 0.f);
      }
    }
    // behavior (K=32)
    {
      v8f acc; float bb = encBb[nb + col];
#pragma unroll
      for (int r = 0; r < 8; ++r) acc[r] = bb;
      acc = wmma_bf(ldA_bf(&s_beh[col][0], 0, hi8),
                    ldA_bf(encBT + (size_t)(nb + col) * 32, 0, hi8), acc);
#pragma unroll
      for (int r = 0; r < 8; ++r) {
        float v = acc[r];
        s_emb[2][r + rsel][nb + col] = (bf16_t)(v > 0.f ? v : 0.f);
      }
    }
  }
  __syncthreads();

  // attention scores: sum_n tanh((emb@W)[.,n]) * v[n]
  for (int m = 0; m < 3; ++m) {
    float sc8[8];
#pragma unroll
    for (int r = 0; r < 8; ++r) sc8[r] = 0.f;
    for (int nt = wv * 4; nt < wv * 4 + 4; ++nt) {
      int nb = nt * 16;
      v8f acc;
#pragma unroll
      for (int r = 0; r < 8; ++r) acc[r] = 0.f;
      const bf16_t* bp = attnWT + (size_t)(nb + col) * EE;
      __builtin_prefetch(attnWT + (size_t)(nb + 16 + col) * EE, 0, 1);
#pragma unroll
      for (int ks = 0; ks < EE; ks += 32)
        acc = wmma_bf(ldA_bf(&s_emb[m][col][0], ks, hi8), ldA_bf(bp, ks, hi8), acc);
      float vn = attnV[nb + col];
#pragma unroll
      for (int r = 0; r < 8; ++r) sc8[r] += tanhf(acc[r]) * vn;
    }
#pragma unroll
    for (int r = 0; r < 8; ++r) atomicAdd(&s_sc[r + rsel][m], sc8[r]);
  }
  __syncthreads();

  // softmax over 3 modalities, then weighted sum of embeddings
  if (tid < 16) {
    float a = s_sc[tid][0], b = s_sc[tid][1], c = s_sc[tid][2];
    float mx = fmaxf(a, fmaxf(b, c));
    float e0 = __expf(a - mx), e1 = __expf(b - mx), e2 = __expf(c - mx);
    float inv = 1.f / (e0 + e1 + e2);
    s_w[tid][0] = e0 * inv; s_w[tid][1] = e1 * inv; s_w[tid][2] = e2 * inv;
  }
  __syncthreads();
  for (int i = tid; i < 16 * EE; i += 128) {
    int r = i >> 8, n = i & 255;
    float v = s_w[r][0] * (float)s_emb[0][r][n] +
              s_w[r][1] * (float)s_emb[1][r][n] +
              s_w[r][2] * (float)s_emb[2][r][n];
    s_wemb[r][n] = (bf16_t)v;
  }
  __syncthreads();

  // fused = relu(wemb @ fuse_W + b), N=512 -> 32 tiles over 4 waves
  for (int nt = wv * 8; nt < wv * 8 + 8; ++nt) {
    int nb = nt * 16;
    v8f acc; float bb = fuseB[nb + col];
#pragma unroll
    for (int r = 0; r < 8; ++r) acc[r] = bb;
    const bf16_t* bp = fuseWT + (size_t)(nb + col) * EE;
    __builtin_prefetch(fuseWT + (size_t)(nb + 16 + col) * EE, 0, 1);
#pragma unroll
    for (int ks = 0; ks < EE; ks += 32)
      acc = wmma_bf(ldA_bf(&s_wemb[col][0], ks, hi8), ldA_bf(bp, ks, hi8), acc);
#pragma unroll
    for (int r = 0; r < 8; ++r) {
      float v = acc[r];
      fused[(size_t)(rowg0 + r + rsel) * DF + nb + col] = v > 0.f ? v : 0.f;
    }
  }
}

// ------------- routed experts: adapter + heads on sorted 16-row tiles ---------
__global__ __launch_bounds__(128) void k_route(
    const float* __restrict__ fused, const int* __restrict__ offs,
    const int* __restrict__ tstart, const int* __restrict__ perm,
    const bf16_t* __restrict__ adW1T, const float* __restrict__ adB1,
    const bf16_t* __restrict__ adW2T, const float* __restrict__ adB2,
    const bf16_t* __restrict__ phWsT, const float* __restrict__ phBs,
    const bf16_t* __restrict__ headT, const float* __restrict__ headB,
    float* __restrict__ out) {
  // 32KB fused(f32) + 16KB (fused bf16 copy, later adapted bf16) + 8KB h/sh + ridx
  __shared__ __align__(16) char smem[16 * DF * 4 + 16 * DF * 2 + 16 * HB * 2 + 64];
  float* s_f = (float*)smem;                                   // [16][512] f32
  bf16_t* s_ad = (bf16_t*)(smem + 16 * DF * 4);                // [16][512] bf16
  bf16_t* s_hs = (bf16_t*)(smem + 16 * DF * 4 + 16 * DF * 2);  // [16][256]
  int* s_ridx = (int*)(smem + 16 * DF * 4 + 16 * DF * 2 + 16 * HB * 2);

  int g = blockIdx.x;
  if (g >= tstart[CC]) return;
  int c = 0;
  while (c < CC - 1 && g >= tstart[c + 1]) ++c;
  int t = g - tstart[c];
  int base = offs[c];
  int ncnt = offs[c + 1] - base;
  int nrows = ncnt - t * 16; if (nrows > 16) nrows = 16;

  const int tid = threadIdx.x;
  const int lane = tid & 31;
  const int wv = tid >> 5;
  const int col = lane & 15;
  const int hi8 = (lane >> 4) * 8;
  const int rsel = (lane >> 4) << 3;

  if (tid < 16) {
    int j = t * 16 + tid;
    s_ridx[tid] = perm[base + (j < ncnt ? j : 0)];
  }
  __syncthreads();
  // gather fused rows: keep f32 (residual) + bf16 copy (stage-1 A operand)
  for (int i = tid; i < 16 * DF; i += 128) {
    int r = i >> 9, n = i & 511;
    float v = fused[(size_t)s_ridx[r] * DF + n];
    s_f[r * DF + n] = v;
    s_ad[r * DF + n] = (bf16_t)v;
  }
  __syncthreads();

  const bf16_t* W1 = adW1T + (size_t)c * HB * DF;   // [256][512]
  const float* b1 = adB1 + c * HB;
  const bf16_t* W2 = adW2T + (size_t)c * DF * HB;   // [512][256]
  const float* b2 = adB2 + c * DF;
  const bf16_t* Ws = phWsT + (size_t)c * HHD * DF;  // [256][512]
  const float* bs = phBs + c * HHD;
  const bf16_t* Wh = headT + (size_t)c * 16 * HHD;  // [16][256]
  const float* bh = headB + c * 16;

  // stage 1: h = relu(fused @ W1 + b1)  (N=256, K=512); A = bf16 fused copy
  for (int nt = wv; nt < 16; nt += 4) {
    int nb = nt * 16;
    v8f acc; float bb = b1[nb + col];
#pragma unroll
    for (int r = 0; r < 8; ++r) acc[r] = bb;
    const bf16_t* bp = W1 + (size_t)(nb + col) * DF;
    __builtin_prefetch(W1 + (size_t)(nb + 64 + col) * DF, 0, 1);
#pragma unroll
    for (int ks = 0; ks < DF; ks += 32)
      acc = wmma_bf(ldA_bf(&s_ad[col * DF], ks, hi8), ldA_bf(bp, ks, hi8), acc);
#pragma unroll
    for (int r = 0; r < 8; ++r) {
      float v = acc[r];
      s_hs[(r + rsel) * HB + nb + col] = (bf16_t)(v > 0.f ? v : 0.f);
    }
  }
  __syncthreads();

  // stage 2: adapted = fused + h @ W2 + b2  (N=512, K=256) — residual in C operand
  // (overwrites the bf16 fused copy in s_ad, which is dead after stage 1)
  for (int nt = wv; nt < 32; nt += 4) {
    int nb = nt * 16;
    v8f acc; float bb = b2[nb + col];
#pragma unroll
    for (int r = 0; r < 8; ++r) acc[r] = bb + s_f[(r + rsel) * DF + nb + col];
    const bf16_t* bp = W2 + (size_t)(nb + col) * HB;
    __builtin_prefetch(W2 + (size_t)(nb + 64 + col) * HB, 0, 1);
#pragma unroll
    for (int ks = 0; ks < HB; ks += 32)
      acc = wmma_bf(ldA_bf(&s_hs[col * HB], ks, hi8), ldA_bf(bp, ks, hi8), acc);
#pragma unroll
    for (int r = 0; r < 8; ++r) s_ad[(r + rsel) * DF + nb + col] = (bf16_t)acc[r];
  }
  __syncthreads();

  // stage 3: sh = relu(adapted @ Ws + bs)  (N=256, K=512)
  for (int nt = wv; nt < 16; nt += 4) {
    int nb = nt * 16;
    v8f acc; float bb = bs[nb + col];
#pragma unroll
    for (int r = 0; r < 8; ++r) acc[r] = bb;
    const bf16_t* bp = Ws + (size_t)(nb + col) * DF;
    __builtin_prefetch(Ws + (size_t)(nb + 64 + col) * DF, 0, 1);
#pragma unroll
    for (int ks = 0; ks < DF; ks += 32)
      acc = wmma_bf(ldA_bf(&s_ad[col * DF], ks, hi8), ldA_bf(bp, ks, hi8), acc);
#pragma unroll
    for (int r = 0; r < 8; ++r) {
      float v = acc[r];
      s_hs[(r + rsel) * HHD + nb + col] = (bf16_t)(v > 0.f ? v : 0.f);
    }
  }
  __syncthreads();

  // stage 4: out = sh @ [Wd|Wr|Wa] + [bd|br|ba]  (N=16 padded, K=256), wave0
  if (wv == 0) {
    v8f acc; float bb = bh[col];
#pragma unroll
    for (int r = 0; r < 8; ++r) acc[r] = bb;
    const bf16_t* bp = Wh + (size_t)col * HHD;
#pragma unroll
    for (int ks = 0; ks < HHD; ks += 32)
      acc = wmma_bf(ldA_bf(&s_hs[col * HHD], ks, hi8), ldA_bf(bp, ks, hi8), acc);
#pragma unroll
    for (int r = 0; r < 8; ++r) {
      int rm = r + rsel;
      if (col < NOUT && rm < nrows)
        out[(size_t)s_ridx[rm] * NOUT + col] = acc[r];
    }
  }
}

// ---------------- host launch ----------------
extern "C" void kernel_launch(void* const* d_in, const int* in_sizes, int n_in,
                              void* d_out, int out_size, void* d_ws,
                              size_t ws_size, hipStream_t stream) {
  const float* price = (const float*)d_in[0];
  const float* sent = (const float*)d_in[1];
  const float* behav = (const float*)d_in[2];
  const int* cid = (const int*)d_in[3];
  const float* enc_pW = (const float*)d_in[4];
  const float* enc_pb = (const float*)d_in[5];
  const float* enc_sW = (const float*)d_in[6];
  const float* enc_sb = (const float*)d_in[7];
  const float* enc_bW = (const float*)d_in[8];
  const float* enc_bb = (const float*)d_in[9];
  const float* attn_W = (const float*)d_in[10];
  const float* attn_v = (const float*)d_in[11];
  const float* fuse_W = (const float*)d_in[12];
  const float* fuse_b = (const float*)d_in[13];
  const float* ad_W1 = (const float*)d_in[14];
  const float* ad_b1 = (const float*)d_in[15];
  const float* ad_W2 = (const float*)d_in[16];
  const float* ad_b2 = (const float*)d_in[17];
  const float* ph_Ws = (const float*)d_in[18];
  const float* ph_bs = (const float*)d_in[19];
  const float* ph_Wd = (const float*)d_in[20];
  const float* ph_bd = (const float*)d_in[21];
  const float* ph_Wr = (const float*)d_in[22];
  const float* ph_br = (const float*)d_in[23];
  const float* ph_Wa = (const float*)d_in[24];
  const float* ph_ba = (const float*)d_in[25];
  float* out = (float*)d_out;
  (void)in_sizes; (void)n_in; (void)out_size; (void)ws_size;

  char* w = (char*)d_ws;
  size_t off = 0;
  auto alloc = [&](size_t bytes) -> char* {
    char* p = w + off;
    off += (bytes + 255) & ~(size_t)255;
    return p;
  };
  float* g_fused = (float*)alloc((size_t)CB * DF * 4);
  bf16_t* encPT = (bf16_t*)alloc((size_t)EE * 32 * 2);
  bf16_t* encST = (bf16_t*)alloc((size_t)EE * 64 * 2);
  bf16_t* encBT = (bf16_t*)alloc((size_t)EE * 32 * 2);
  bf16_t* attnWT = (bf16_t*)alloc((size_t)EE * EE * 2);
  bf16_t* fuseWT = (bf16_t*)alloc((size_t)DF * EE * 2);
  bf16_t* adW1T = (bf16_t*)alloc((size_t)CC * HB * DF * 2);
  bf16_t* adW2T = (bf16_t*)alloc((size_t)CC * DF * HB * 2);
  bf16_t* phWsT = (bf16_t*)alloc((size_t)CC * HHD * DF * 2);
  bf16_t* headT = (bf16_t*)alloc((size_t)CC * 16 * HHD * 2);
  float* headB = (float*)alloc((size_t)CC * 16 * 4);
  int* counts = (int*)alloc(CC * 4);
  int* cursor = (int*)alloc(CC * 4);
  int* offs = (int*)alloc((CC + 1) * 4);
  int* tstart = (int*)alloc((CC + 1) * 4);
  int* perm = (int*)alloc((size_t)CB * 4);

  // weight prep (bf16, transposed to [N][Kpad])
  k_convT<<<256, 256, 0, stream>>>(encPT, enc_pW, 1, FP, EE, 32);
  k_convT<<<256, 256, 0, stream>>>(encST, enc_sW, 1, DSNT, EE, 64);
  k_convT<<<256, 256, 0, stream>>>(encBT, enc_bW, 1, DBH, EE, 32);
  k_convT<<<512, 256, 0, stream>>>(attnWT, attn_W, 1, EE, EE, EE);
  k_convT<<<512, 256, 0, stream>>>(fuseWT, fuse_W, 1, EE, DF, EE);
  k_convT<<<2048, 256, 0, stream>>>(adW1T, ad_W1, CC, DF, HB, DF);
  k_convT<<<2048, 256, 0, stream>>>(adW2T, ad_W2, CC, HB, DF, HB);
  k_convT<<<2048, 256, 0, stream>>>(phWsT, ph_Ws, CC, DF, HHD, DF);
  k_headpack<<<128, 256, 0, stream>>>(headT, headB, ph_Wd, ph_bd, ph_Wr, ph_br,
                                      ph_Wa, ph_ba);

  // counting sort by client id
  k_zero<<<1, 32, 0, stream>>>(counts, cursor);
  k_count<<<CB / 256, 256, 0, stream>>>(cid, counts);
  k_scan<<<1, 1, 0, stream>>>(counts, offs, tstart);
  k_place<<<CB / 256, 256, 0, stream>>>(cid, offs, cursor, perm);

  // trunk + routed experts
  k_encode_fuse<<<CB / 16, 128, 0, stream>>>(
      price, sent, behav, encPT, enc_pb, encST, enc_sb, encBT, enc_bb, attnWT,
      attn_v, fuseWT, fuse_b, g_fused);
  k_route<<<CB / 16 + CC, 128, 0, stream>>>(g_fused, offs, tstart, perm, adW1T,
                                            ad_b1, adW2T, ad_b2, phWsT, ph_bs,
                                            headT, headB, out);
}
